// RetinaFace_12008728560058
// MI455X (gfx1250) — matrix operand order repaired and checked by reading, not compiled, not run on hardware
//
#include <hip/hip_runtime.h>

// ---------------------------------------------------------------------------
// RetinaFace forward for MI455X (gfx1250): f16-NHWC implicit-GEMM convs on
// v_wmma_f32_16x16x32_f16, fused BN/bias/ReLU epilogues, fused head+softmax+
// decode, radix-select top-k + bitonic tie sort + greedy NMS.
//
// Round 2: conv kernels templated on CIN (all powers of two) so K->(kidx,cin)
// decomposition is shifts/masks instead of runtime integer division, Ktot and
// chunk count are constexpr, and the K loop fully unrolls (constant-folded
// padding checks, loads hoisted ahead of the WMMA chain).
// ---------------------------------------------------------------------------

typedef __attribute__((ext_vector_type(16))) _Float16 v16h;
typedef __attribute__((ext_vector_type(8)))  _Float16 v8h;
typedef __attribute__((ext_vector_type(8)))  float    v8f;

#define IMGSZ 1280
#define NPRI  67200
#define TOPKN 1000
#define CANDN 2048
#define BATCH 4

// ------------------------------ small utils -------------------------------

__device__ inline v8h zero_v8h() {
    v8h z;
#pragma unroll
    for (int i = 0; i < 8; ++i) z[i] = (_Float16)0.0f;
    return z;
}

// ------------------------- layout transform kernels -----------------------

// f32 NCHW -> f16 NHWC
__global__ void nchw2nhwc_f16(const float* __restrict__ in, _Float16* __restrict__ out,
                              int Bn, int C, int H, int W) {
    int idx = blockIdx.x * blockDim.x + threadIdx.x;
    int tot = Bn * C * H * W;
    if (idx >= tot) return;
    int w = idx % W;
    int t = idx / W;
    int h = t % H; t /= H;
    int c = t % C;
    int b = t / C;
    out[(((size_t)b * H + h) * W + w) * C + c] = (_Float16)in[idx];
}

// P_hi[b,h,w,c] += P_lo[b,h/2,w/2,c]   (nearest 2x upsample add), f16 NHWC
__global__ void upsample_add_f16(_Float16* __restrict__ hi, const _Float16* __restrict__ lo,
                                 int Bn, int H, int W, int C) {
    int idx = blockIdx.x * blockDim.x + threadIdx.x;
    int tot = Bn * H * W * C;
    if (idx >= tot) return;
    int c = idx % C;
    int t = idx / C;
    int w = t % W; t /= W;
    int h = t % H;
    int b = t / H;
    size_t li = (((size_t)b * (H >> 1) + (h >> 1)) * (W >> 1) + (w >> 1)) * C + c;
    hi[idx] = (_Float16)((float)hi[idx] + (float)lo[li]);
}

// Pack OIHW f32 weights (times per-channel BN scale) into the conv kernel's contiguous-k wave32
// WMMA B-fragment layout: dst[chunk][ntile][lane(32)][e(16)] with
//   n = ntile*16 + (lane&15),  k = chunk*32 + (lane>>4)*16 + e,
//   k = (kh*KK+kw)*Cin + cin, zero-padded past Ktot.
__global__ void pack_w(const float* __restrict__ wsrc, const float* __restrict__ ssrc,
                       _Float16* __restrict__ dst, int Cin, int Cout, int KK, int nChunks) {
    int idx = blockIdx.x * blockDim.x + threadIdx.x;
    int NT = Cout >> 4;
    int total = nChunks * NT * 512;
    if (idx >= total) return;
    int e    = idx & 15;
    int lane = (idx >> 4) & 31;
    int t    = idx >> 9;         // chunk*NT + nt
    int nt = t % NT;
    int c  = t / NT;
    int hl = lane >> 4, lr = lane & 15;
    int n = nt * 16 + lr;
    int k = c * 32 + hl * 16 + e;
    int Ktot = Cin * KK * KK;
    float v = 0.0f;
    if (k < Ktot) {
        int kidx = k / Cin, cin = k % Cin;
        int kh = kidx / KK, kw = kidx % KK;
        v = wsrc[(((size_t)n * Cin + cin) * KK + kh) * KK + kw];
        if (ssrc) v *= ssrc[n];
    }
    dst[idx] = (_Float16)v;
}

// --------------------------- WMMA conv kernel ------------------------------

// Load one 8-wide contiguous-channel K-run of the A fragment (f16 NHWC input).
// CIN is a compile-time power of two: kidx/cin0 become shift/mask; the K-pad
// check constant-folds under full unroll; 1x1 convs have no spatial check.
template <int KK, int CIN>
__device__ inline v8h load_a_run(const _Float16* __restrict__ in, int k0,
                                 int b, int h, int w, int H, int W, int pad) {
    constexpr int KTOT = CIN * KK * KK;
    if (k0 >= KTOT) return zero_v8h();          // compile-time per unrolled chunk
    int kidx = k0 / CIN;                        // shift (CIN = power of two)
    int cin0 = k0 & (CIN - 1);                  // mask
    int h2 = h, w2 = w;
    if (KK == 3) {
        h2 += kidx / 3 - pad;                   // kidx constant under unroll
        w2 += kidx % 3 - pad;
        if ((unsigned)h2 >= (unsigned)H || (unsigned)w2 >= (unsigned)W) return zero_v8h();
    }
    size_t idx = ((size_t)((b * H + h2) * W + w2)) * (size_t)CIN + cin0;
    return *(const v8h*)(in + idx);
}

// One wave computes a 16-pixel x (NT*16)-channel tile. K loop in chunks of 32,
// fully unrolled (constexpr trip count). Epilogue: +bias, optional ReLU, f16
// NHWC store at channel offset `coff` with stride `Cstride` (free SSH concat).
template <int NT, int KK, int CIN>
__global__ __launch_bounds__(256) void conv_wmma_kernel(
    const _Float16* __restrict__ in, const _Float16* __restrict__ wp,
    const float* __restrict__ bias, _Float16* __restrict__ out,
    int H, int W, int Cstride, int coff, int pad, int relu, int Mtiles) {
    constexpr int NCHUNKS = (CIN * KK * KK + 31) / 32;

    int gid   = blockIdx.x * blockDim.x + threadIdx.x;
    int mtile = gid >> 5;
    int lane  = threadIdx.x & 31;
    if (mtile >= Mtiles) return;
    int hl = lane >> 4, lr = lane & 15;

    int HW  = H * W;
    int p   = mtile * 16 + lr;      // this lane's A-matrix row (pixel)
    int b   = p / HW;
    int rem = p - b * HW;
    int h   = rem / W;
    int w   = rem - h * W;

    v8f acc[NT];
#pragma unroll
    for (int n = 0; n < NT; ++n)
#pragma unroll
        for (int q = 0; q < 8; ++q) acc[n][q] = 0.0f;

#pragma unroll
    for (int c = 0; c < NCHUNKS; ++c) {
        // A fragment: lanes 0-15 hold K {0..15}+c*32, lanes 16-31 {16..31}+c*32
        int kbase = c * 32 + hl * 16;
        v8h lo = load_a_run<KK, CIN>(in, kbase,      b, h, w, H, W, pad);
        v8h hi = load_a_run<KK, CIN>(in, kbase + 8,  b, h, w, H, W, pad);
        v16h a;
#pragma unroll
        for (int q = 0; q < 8; ++q) { a[q] = lo[q]; a[q + 8] = hi[q]; }

#pragma unroll
        for (int n = 0; n < NT; ++n) {
            const _Float16* wpc = wp + (((size_t)(c * NT + n)) << 9) + ((size_t)lane << 4);
            const v8h* bp = (const v8h*)wpc;
            v8h b0 = bp[0], b1 = bp[1];
            v16h bv;
#pragma unroll
            for (int q = 0; q < 8; ++q) { bv[q] = b0[q]; bv[q + 8] = b1[q]; }
            acc[n] = __builtin_amdgcn_wmma_f32_16x16x32_f16(
                false, a, false, bv, (short)0, acc[n], false, false);
        }
    }

    // D layout: lane holds column n = lr; VGPR r is row m = r + hl*8.
#pragma unroll
    for (int n = 0; n < NT; ++n) {
        int ch = n * 16 + lr;
        float bb = bias[ch];
#pragma unroll
        for (int r = 0; r < 8; ++r) {
            int m = hl * 8 + r;
            float v = acc[n][r] + bb;
            if (relu) v = v > 0.0f ? v : 0.0f;
            size_t pp = (size_t)(mtile * 16 + m);
            out[pp * (size_t)Cstride + coff + ch] = (_Float16)v;
        }
    }
}

// --------------------- heads + softmax + prior decode ----------------------

__global__ void head_decode_kernel(const _Float16* __restrict__ X,
                                   const float* __restrict__ wb, const float* __restrict__ bb,
                                   const float* __restrict__ wc, const float* __restrict__ bc,
                                   float* __restrict__ boxes, float* __restrict__ scores,
                                   int Bn, int H, int W, int stride, float s0, float s1,
                                   int priorOff) {
    int idx = blockIdx.x * blockDim.x + threadIdx.x;
    int tot = Bn * H * W;
    if (idx >= tot) return;
    int HW  = H * W;
    int b   = idx / HW;
    int rem = idx - b * HW;
    int h   = rem / W;
    int w   = rem - h * W;
    const _Float16* x = X + (size_t)idx * 64;

    float lb[8], lc[4];
#pragma unroll
    for (int o = 0; o < 8; ++o) lb[o] = bb[o];
#pragma unroll
    for (int o = 0; o < 4; ++o) lc[o] = bc[o];
    for (int i = 0; i < 64; ++i) {
        float xv = (float)x[i];
#pragma unroll
        for (int o = 0; o < 8; ++o) lb[o] += wb[o * 64 + i] * xv;
#pragma unroll
        for (int o = 0; o < 4; ++o) lc[o] += wc[o * 64 + i] * xv;
    }
    const float inv = 1.0f / (float)IMGSZ;
#pragma unroll
    for (int a2 = 0; a2 < 2; ++a2) {
        float sc = 1.0f / (1.0f + expf(lc[2 * a2] - lc[2 * a2 + 1]));  // softmax[:,1]
        float cx = (w + 0.5f) * stride * inv;
        float cy = (h + 0.5f) * stride * inv;
        float pw = (a2 ? s1 : s0) * inv, ph = pw;
        float gx = cx + lb[4 * a2 + 0] * 0.1f * pw;
        float gy = cy + lb[4 * a2 + 1] * 0.1f * ph;
        float gw = pw * expf(lb[4 * a2 + 2] * 0.2f);
        float gh = ph * expf(lb[4 * a2 + 3] * 0.2f);
        int prior = priorOff + rem * 2 + a2;
        size_t o4 = (size_t)b * NPRI + prior;
        float* ob = boxes + o4 * 4;
        ob[0] = gx - 0.5f * gw; ob[1] = gy - 0.5f * gh;
        ob[2] = gx + 0.5f * gw; ob[3] = gy + 0.5f * gh;
        scores[o4] = sc;
    }
}

// ---------------- per-image top-k + tie-ordered sort + NMS -----------------

__global__ __launch_bounds__(1024) void select_nms_kernel(
    const float* __restrict__ boxesAll, const float* __restrict__ scoresAll,
    float* __restrict__ outB, float* __restrict__ outS, float* __restrict__ outK,
    unsigned* __restrict__ candWs) {
    __shared__ unsigned hist[256];
    __shared__ unsigned prefix_s, need_s, cntG, cntE;
    __shared__ unsigned long long keys[CANDN];
    __shared__ float bx0[TOPKN], by0[TOPKN], bx1[TOPKN], by1[TOPKN], scl[TOPKN];
    __shared__ unsigned char keep[TOPKN];

    int b = blockIdx.x, tid = threadIdx.x;
    const float* S  = scoresAll + (size_t)b * NPRI;
    const float* BX = boxesAll + (size_t)b * NPRI * 4;
    unsigned* cand  = candWs + (size_t)b * CANDN;

    // --- radix-select the TOPKN-th largest score (scores >= 0 -> bit order) ---
    if (tid == 0) { prefix_s = 0u; need_s = TOPKN; }
    __syncthreads();
    for (int pass = 0; pass < 4; ++pass) {
        int shift = 24 - pass * 8;
        if (tid < 256) hist[tid] = 0u;
        __syncthreads();
        unsigned pmask = (pass == 0) ? 0u : (0xFFFFFFFFu << (shift + 8));
        unsigned pref  = prefix_s;
        for (int i = tid; i < NPRI; i += 1024) {
            unsigned u = __float_as_uint(S[i]);
            if ((u & pmask) == pref) atomicAdd(&hist[(u >> shift) & 0xFFu], 1u);
        }
        __syncthreads();
        if (tid == 0) {
            unsigned needL = need_s, cum = 0u;
            int d = 255;
            for (; d > 0; --d) {
                unsigned c2 = hist[d];
                if (cum + c2 >= needL) break;
                cum += c2;
            }
            prefix_s = pref | ((unsigned)d << shift);
            need_s   = needL - cum;
        }
        __syncthreads();
    }
    unsigned Tbits = prefix_s;

    // --- compact candidates: all strictly-greater, then ties (capped) ---
    if (tid == 0) { cntG = 0u; cntE = 0u; }
    __syncthreads();
    for (int i = tid; i < NPRI; i += 1024) {
        unsigned u = __float_as_uint(S[i]);
        if (u > Tbits) {
            unsigned p2 = atomicAdd(&cntG, 1u);
            if (p2 < CANDN) cand[p2] = (unsigned)i;
        }
    }
    __syncthreads();
    unsigned nG = cntG; if (nG > CANDN) nG = CANDN;
    for (int i = tid; i < NPRI; i += 1024) {
        unsigned u = __float_as_uint(S[i]);
        if (u == Tbits) {
            unsigned p2 = nG + atomicAdd(&cntE, 1u);
            if (p2 < CANDN) cand[p2] = (unsigned)i;
        }
    }
    __syncthreads();
    unsigned nC = nG + cntE; if (nC > CANDN) nC = CANDN;

    // --- bitonic sort descending on (scoreBits, ~index) to match jax top_k ---
    for (int i = tid; i < CANDN; i += 1024) {
        unsigned long long key = 0ull;
        if (i < (int)nC) {
            unsigned id = cand[i];
            unsigned u  = __float_as_uint(S[id]);
            key = ((unsigned long long)u << 32) | (unsigned)(~id);
        }
        keys[i] = key;
    }
    __syncthreads();
    for (int k = 2; k <= CANDN; k <<= 1) {
        for (int j = k >> 1; j > 0; j >>= 1) {
            for (int i = tid; i < CANDN; i += 1024) {
                int ixj = i ^ j;
                if (ixj > i) {
                    unsigned long long a = keys[i], c2 = keys[ixj];
                    bool dir = ((i & k) == 0);
                    if ((a < c2) == dir) { keys[i] = c2; keys[ixj] = a; }
                }
            }
            __syncthreads();
        }
    }

    // --- gather top-1000 boxes, write boxes/scores, init keep ---
    for (int r = tid; r < TOPKN; r += 1024) {
        unsigned long long key = keys[r];
        float s = __uint_as_float((unsigned)(key >> 32));
        unsigned id = ~((unsigned)(key & 0xFFFFFFFFull));
        float x0 = 0.f, y0 = 0.f, x1 = 0.f, y1 = 0.f;
        if (key != 0ull && id < NPRI) {
            const float* bo = BX + (size_t)id * 4;
            x0 = bo[0]; y0 = bo[1]; x1 = bo[2]; y1 = bo[3];
        } else {
            s = 0.f;
        }
        bx0[r] = x0; by0[r] = y0; bx1[r] = x1; by1[r] = y1; scl[r] = s;
        keep[r] = (s > 0.02f) ? 1 : 0;
        float* ob = outB + ((size_t)b * TOPKN + r) * 4;
        ob[0] = x0; ob[1] = y0; ob[2] = x1; ob[3] = y1;
        outS[(size_t)b * TOPKN + r] = s;
    }
    __syncthreads();

    // --- greedy NMS: serial over i, parallel suppression over j ---
    for (int i = 0; i < TOPKN - 1; ++i) {
        if (keep[i]) {
            float ax0 = bx0[i], ay0 = by0[i], ax1 = bx1[i], ay1 = by1[i];
            float aA = (ax1 - ax0) * (ay1 - ay0);
            for (int r = tid; r < TOPKN; r += 1024) {
                if (r > i && keep[r]) {
                    float ix0 = fmaxf(ax0, bx0[r]), iy0 = fmaxf(ay0, by0[r]);
                    float ix1 = fminf(ax1, bx1[r]), iy1 = fminf(ay1, by1[r]);
                    float iw = fmaxf(ix1 - ix0, 0.f), ih = fmaxf(iy1 - iy0, 0.f);
                    float inter = iw * ih;
                    float bA = (bx1[r] - bx0[r]) * (by1[r] - by0[r]);
                    float iou = inter / (aA + bA - inter + 1e-9f);
                    if (iou > 0.4f) keep[r] = 0;
                }
            }
        }
        __syncthreads();
    }
    for (int r = tid; r < TOPKN; r += 1024)
        outK[(size_t)b * TOPKN + r] = keep[r] ? 1.0f : 0.0f;
}

// ------------------------------- host side ---------------------------------

extern "C" void kernel_launch(void* const* d_in, const int* in_sizes, int n_in,
                              void* d_out, int out_size, void* d_ws, size_t ws_size,
                              hipStream_t stream) {
    (void)in_sizes; (void)n_in; (void)out_size; (void)ws_size;

    // Input map (jax pytree = sorted dict keys):
    // 0..2 feats; params: head_box[3]{b,w}, head_cls[3]{b,w},
    // lat[3]{b,s,w}, smooth[2]{b,s,w}, ssh[3]{c1..c5}{b,s,w}
    const float* featsF32[3] = {(const float*)d_in[0], (const float*)d_in[1], (const float*)d_in[2]};
    const float *hb_b[3], *hb_w[3], *hc_b[3], *hc_w[3];
    for (int i = 0; i < 3; ++i) {
        hb_b[i] = (const float*)d_in[3 + i * 2];
        hb_w[i] = (const float*)d_in[4 + i * 2];
        hc_b[i] = (const float*)d_in[9 + i * 2];
        hc_w[i] = (const float*)d_in[10 + i * 2];
    }
    const float *lat_b[3], *lat_s[3], *lat_w[3];
    for (int i = 0; i < 3; ++i) {
        lat_b[i] = (const float*)d_in[15 + i * 3];
        lat_s[i] = (const float*)d_in[16 + i * 3];
        lat_w[i] = (const float*)d_in[17 + i * 3];
    }
    const float *sm_b[2], *sm_s[2], *sm_w[2];
    for (int i = 0; i < 2; ++i) {
        sm_b[i] = (const float*)d_in[24 + i * 3];
        sm_s[i] = (const float*)d_in[25 + i * 3];
        sm_w[i] = (const float*)d_in[26 + i * 3];
    }
    const float *ssh_b[3][5], *ssh_s[3][5], *ssh_w[3][5];
    for (int i = 0; i < 3; ++i)
        for (int c = 0; c < 5; ++c) {
            int base2 = 30 + i * 15 + c * 3;
            ssh_b[i][c] = (const float*)d_in[base2 + 0];
            ssh_s[i][c] = (const float*)d_in[base2 + 1];
            ssh_w[i][c] = (const float*)d_in[base2 + 2];
        }

    const int HH[3]   = {160, 80, 40};
    const int CINS[3] = {64, 128, 256};
    const int STR[3]  = {8, 16, 32};
    const float SZ0[3] = {16.f, 64.f, 256.f};
    const float SZ1[3] = {32.f, 128.f, 512.f};
    const int POFF[3] = {0, 51200, 64000};
    int pix[3];
    for (int i = 0; i < 3; ++i) pix[i] = BATCH * HH[i] * HH[i];

    // ---- workspace bump allocator (all offsets 256B aligned) ----
    char* wsbase = (char*)d_ws;
    size_t off = 0;
    auto take = [&](size_t bytes) -> char* {
        char* p = wsbase + off;
        off += (bytes + 255) & ~(size_t)255;
        return p;
    };
    _Float16* inF16[3];
    for (int i = 0; i < 3; ++i) inF16[i] = (_Float16*)take((size_t)pix[i] * CINS[i] * 2);
    _Float16* P[3];
    for (int i = 0; i < 3; ++i) P[i] = (_Float16*)take((size_t)pix[i] * 64 * 2);
    _Float16* Xs[2];
    for (int i = 0; i < 2; ++i) Xs[i] = (_Float16*)take((size_t)pix[i] * 64 * 2);
    _Float16 *tbuf[3], *t2buf[3];
    for (int i = 0; i < 3; ++i) tbuf[i]  = (_Float16*)take((size_t)pix[i] * 16 * 2);
    for (int i = 0; i < 3; ++i) t2buf[i] = (_Float16*)take((size_t)pix[i] * 16 * 2);
    // SSH concat outputs alias converted inputs (dead after lateral convs; Cin>=64)
    _Float16* F[3] = {inF16[0], inF16[1], inF16[2]};

    auto packHalves = [](int Cin, int Cout, int KK) -> size_t {
        int nC = (Cin * KK * KK + 31) / 32;
        return (size_t)nC * (Cout / 16) * 512;
    };
    _Float16* wp_lat[3];
    for (int i = 0; i < 3; ++i) wp_lat[i] = (_Float16*)take(packHalves(CINS[i], 64, 1) * 2);
    _Float16* wp_sm[2];
    for (int i = 0; i < 2; ++i) wp_sm[i] = (_Float16*)take(packHalves(64, 64, 3) * 2);
    const int sshCin[5]  = {64, 64, 16, 16, 16};
    const int sshCout[5] = {32, 16, 16, 16, 16};
    _Float16* wp_ssh[3][5];
    for (int i = 0; i < 3; ++i)
        for (int c = 0; c < 5; ++c)
            wp_ssh[i][c] = (_Float16*)take(packHalves(sshCin[c], sshCout[c], 3) * 2);
    float* boxesBuf  = (float*)take((size_t)BATCH * NPRI * 4 * sizeof(float));
    float* scoresBuf = (float*)take((size_t)BATCH * NPRI * sizeof(float));
    unsigned* candBuf = (unsigned*)take((size_t)BATCH * CANDN * sizeof(unsigned));

    // ---- launch helpers ----
    auto packLaunch = [&](const float* w, const float* s, _Float16* dst, int Cin, int Cout, int KK) {
        int nChunks = (Cin * KK * KK + 31) / 32;
        int total = nChunks * (Cout / 16) * 512;
        pack_w<<<(total + 255) / 256, 256, 0, stream>>>(w, s, dst, Cin, Cout, KK, nChunks);
    };
    auto conv = [&](const _Float16* in, const _Float16* wp, const float* bias, _Float16* out,
                    int H, int W, int Cin, int Cout, int Cstride, int coff, int KK, int pad,
                    int relu) {
        int Mtiles = (BATCH * H * W) / 16;
        int blocks = (Mtiles * 32 + 255) / 256;
        int NT = Cout / 16;
        dim3 g(blocks), t(256);
        if (KK == 1) {  // lateral 1x1, NT==4, Cin in {64,128,256}
            if (Cin == 64)
                conv_wmma_kernel<4, 1, 64><<<g, t, 0, stream>>>(in, wp, bias, out, H, W, Cstride, coff, pad, relu, Mtiles);
            else if (Cin == 128)
                conv_wmma_kernel<4, 1, 128><<<g, t, 0, stream>>>(in, wp, bias, out, H, W, Cstride, coff, pad, relu, Mtiles);
            else
                conv_wmma_kernel<4, 1, 256><<<g, t, 0, stream>>>(in, wp, bias, out, H, W, Cstride, coff, pad, relu, Mtiles);
        } else if (Cin == 64) {  // smooth / ssh c1 / ssh c2
            if (NT == 4)
                conv_wmma_kernel<4, 3, 64><<<g, t, 0, stream>>>(in, wp, bias, out, H, W, Cstride, coff, pad, relu, Mtiles);
            else if (NT == 2)
                conv_wmma_kernel<2, 3, 64><<<g, t, 0, stream>>>(in, wp, bias, out, H, W, Cstride, coff, pad, relu, Mtiles);
            else
                conv_wmma_kernel<1, 3, 64><<<g, t, 0, stream>>>(in, wp, bias, out, H, W, Cstride, coff, pad, relu, Mtiles);
        } else {  // ssh c3/c4/c5: Cin==16, NT==1
            conv_wmma_kernel<1, 3, 16><<<g, t, 0, stream>>>(in, wp, bias, out, H, W, Cstride, coff, pad, relu, Mtiles);
        }
    };

    // ---- 1) convert inputs f32 NCHW -> f16 NHWC ----
    for (int i = 0; i < 3; ++i) {
        int tot = pix[i] * CINS[i];
        nchw2nhwc_f16<<<(tot + 255) / 256, 256, 0, stream>>>(featsF32[i], inF16[i], BATCH, CINS[i], HH[i], HH[i]);
    }

    // ---- 2) pack all conv weights (BN scale folded in) ----
    for (int i = 0; i < 3; ++i) packLaunch(lat_w[i], lat_s[i], wp_lat[i], CINS[i], 64, 1);
    for (int i = 0; i < 2; ++i) packLaunch(sm_w[i], sm_s[i], wp_sm[i], 64, 64, 3);
    for (int i = 0; i < 3; ++i)
        for (int c = 0; c < 5; ++c)
            packLaunch(ssh_w[i][c], ssh_s[i][c], wp_ssh[i][c], sshCin[c], sshCout[c], 3);

    // ---- 3) lateral 1x1 convs (+ReLU) -> P ----
    for (int i = 0; i < 3; ++i)
        conv(inF16[i], wp_lat[i], lat_b[i], P[i], HH[i], HH[i], CINS[i], 64, 64, 0, 1, 0, 1);

    // ---- 4) top-down merge (nearest 2x upsample add) ----
    {
        int tot1 = pix[1] * 64;
        upsample_add_f16<<<(tot1 + 255) / 256, 256, 0, stream>>>(P[1], P[2], BATCH, HH[1], HH[1], 64);
        int tot0 = pix[0] * 64;
        upsample_add_f16<<<(tot0 + 255) / 256, 256, 0, stream>>>(P[0], P[1], BATCH, HH[0], HH[0], 64);
    }

    // ---- 5) smooth 3x3 (+ReLU) on P3, P4 only ----
    for (int i = 0; i < 2; ++i)
        conv(P[i], wp_sm[i], sm_b[i], Xs[i], HH[i], HH[i], 64, 64, 64, 0, 3, 1, 1);
    const _Float16* Xin[3] = {Xs[0], Xs[1], P[2]};

    // ---- 6) SSH per level; final relu folded into branch convs; concat via coff ----
    for (int i = 0; i < 3; ++i) {
        int H = HH[i];
        conv(Xin[i],   wp_ssh[i][0], ssh_b[i][0], F[i],     H, H, 64, 32, 64, 0,  3, 1, 1); // c1 -> ch 0..31
        conv(Xin[i],   wp_ssh[i][1], ssh_b[i][1], tbuf[i],  H, H, 64, 16, 16, 0,  3, 1, 1); // c2 (relu)
        conv(tbuf[i],  wp_ssh[i][2], ssh_b[i][2], F[i],     H, H, 16, 16, 64, 32, 3, 1, 1); // c3 -> ch 32..47
        conv(tbuf[i],  wp_ssh[i][3], ssh_b[i][3], t2buf[i], H, H, 16, 16, 16, 0,  3, 1, 1); // c4 (relu)
        conv(t2buf[i], wp_ssh[i][4], ssh_b[i][4], F[i],     H, H, 16, 16, 64, 48, 3, 1, 1); // c5 -> ch 48..63
    }

    // ---- 7) heads + softmax + prior decode ----
    for (int i = 0; i < 3; ++i) {
        int tot = pix[i];
        head_decode_kernel<<<(tot + 127) / 128, 128, 0, stream>>>(
            F[i], hb_w[i], hb_b[i], hc_w[i], hc_b[i], boxesBuf, scoresBuf,
            BATCH, HH[i], HH[i], STR[i], SZ0[i], SZ1[i], POFF[i]);
    }

    // ---- 8) per-image top-k + NMS ----
    float* outB = (float*)d_out;                     // [B,1000,4]
    float* outS = outB + (size_t)BATCH * TOPKN * 4;  // [B,1000]
    float* outK = outS + (size_t)BATCH * TOPKN;      // [B,1000]
    select_nms_kernel<<<BATCH, 1024, 0, stream>>>(boxesBuf, scoresBuf, outB, outS, outK, candBuf);
}
